// MultiScalePeriodDetector_34746285425256
// MI455X (gfx1250) — matrix operand order repaired and checked by reading, not compiled, not run on hardware
//
#include <hip/hip_runtime.h>
#include <cstdint>
#include <cstddef>

typedef __attribute__((ext_vector_type(16))) _Float16 v16h;
typedef __attribute__((ext_vector_type(8)))  float    v8f;
typedef __attribute__((ext_vector_type(4)))  int      v4i;

#define WG_B   32
#define SEQ_T  2048
#define CH_C   64
#define NSIG   2048      /* B*C */
#define TOPK   5
#define TC_W   1027      /* wavelet cA length                  */
#define KW_P   1056      /* TC_W padded to K%32==0             */
#define FC_W   514       /* rfft bins of cA                    */
#define NW_P   576       /* FC_W padded to N%64==0             */
#define F_R    1025      /* rfft bins of T=2048                */
#define NS_P   1088      /* F_R padded to N%64==0              */

#if __has_builtin(__builtin_amdgcn_global_load_async_to_lds_b128)
#define HAS_ASYNC_LDS 1
#else
#define HAS_ASYNC_LDS 0
#endif

__constant__ float DB4[8] = {
  -0.010597401784997278f, 0.032883011666982945f, 0.030841381835986965f,
  -0.18703481171888114f, -0.02798376941698385f, 0.6308807679295904f,
  0.7148465705525415f,   0.23037781330885523f };

__device__ inline float hash01(int n, int k) {
  unsigned u = (unsigned)n * 1664525u + (unsigned)k * 1013904223u + 0x9E3779B9u;
  u ^= u >> 13; u *= 2654435761u; u ^= u >> 16;
  return (float)(u & 0x00FFFFFFu) * (1.0f / 16777216.0f);
}

union F16x16 { uint4 q[2]; v16h v; };

__device__ inline void stage16(const _Float16* __restrict__ g, _Float16* l) {
#if HAS_ASYNC_LDS
  __builtin_amdgcn_global_load_async_to_lds_b128(
      (__attribute__((address_space(1))) v4i*)g,
      (__attribute__((address_space(3))) v4i*)l, 0, 0);
#else
  *reinterpret_cast<uint4*>(l) = *reinterpret_cast<const uint4*>(g);
#endif
}

__device__ inline void stage_wait() {
#if HAS_ASYNC_LDS
#if __has_builtin(__builtin_amdgcn_s_wait_asynccnt)
  __builtin_amdgcn_s_wait_asynccnt(0);
#else
  asm volatile("s_wait_asynccnt 0x0" ::: "memory");
#endif
#endif
}

/* --------- WMMA f16 GEMM: C = alpha*A(MxK)*BT(NxK)^T [+C] ----------------- */
/* Requires M%64==0, N%64==0, K%32==0.  B is passed TRANSPOSED (row n = out   */
/* column), so both A and B tiles stage with coalesced b128 (async direct-to- */
/* LDS when available) and both WMMA fragments load as 2x b128 from LDS.      */
/* 8 waves: 4(M) x 2(N), each wave owns 16 rows x 32 cols = two               */
/* v_wmma_f32_16x16x32_f16 accumulators.                                      */
__global__ __launch_bounds__(256)
void k_wmma_gemm(const _Float16* __restrict__ A, int lda,
                 const _Float16* __restrict__ BT, int ldbt,
                 float* __restrict__ C, int ldc,
                 int M, int N, int K, float alpha, int accumulate)
{
  __shared__ _Float16 As [64][40];   /* 64 rows(m)  x 32 k */
  __shared__ _Float16 BsT[64][40];   /* 64 rows(n)  x 32 k */
  const int tid  = threadIdx.x;
  const int lane = tid & 31;
  const int wave = tid >> 5;
  const int wm   = wave & 3;
  const int wn   = wave >> 2;
  const int hi   = lane >> 4;
  const int lo   = lane & 15;
  const int rowBase = blockIdx.y * 64;
  const int colBase = blockIdx.x * 64;

  const int sR = tid >> 2;           /* 0..63 staging row  */
  const int sC = (tid & 3) * 8;      /* 0,8,16,24          */
  const _Float16* aBase = A  + (size_t)(rowBase + sR) * lda  + sC;
  const _Float16* bBase = BT + (size_t)(colBase + sR) * ldbt + sC;

  v8f acc0 = {}; v8f acc1 = {};

  for (int k0 = 0; k0 < K; k0 += 32) {
    stage16(aBase + k0, &As [sR][sC]);
    stage16(bBase + k0, &BsT[sR][sC]);
    if (k0 + 32 < K) {
      __builtin_prefetch(aBase + k0 + 32, 0, 1);
      __builtin_prefetch(bBase + k0 + 32, 0, 1);
    }
    stage_wait();
    __syncthreads();

    F16x16 fa, fb0, fb1;
    const int am = wm * 16 + lo;
    fa.q[0]  = *reinterpret_cast<const uint4*>(&As[am][hi * 16]);
    fa.q[1]  = *reinterpret_cast<const uint4*>(&As[am][hi * 16 + 8]);
    const int bn0 = wn * 32 + lo;
    fb0.q[0] = *reinterpret_cast<const uint4*>(&BsT[bn0][hi * 16]);
    fb0.q[1] = *reinterpret_cast<const uint4*>(&BsT[bn0][hi * 16 + 8]);
    fb1.q[0] = *reinterpret_cast<const uint4*>(&BsT[bn0 + 16][hi * 16]);
    fb1.q[1] = *reinterpret_cast<const uint4*>(&BsT[bn0 + 16][hi * 16 + 8]);

    acc0 = __builtin_amdgcn_wmma_f32_16x16x32_f16(false, fa.v, false, fb0.v,
                                                  (short)0, acc0, false, false);
    acc1 = __builtin_amdgcn_wmma_f32_16x16x32_f16(false, fa.v, false, fb1.v,
                                                  (short)0, acc1, false, false);
    __syncthreads();
  }

  const int gr0 = rowBase + wm * 16 + hi * 8;
  const int gc0 = colBase + wn * 32 + lo;
  #pragma unroll
  for (int r = 0; r < 8; ++r) {
    size_t o0 = (size_t)(gr0 + r) * ldc + gc0;
    size_t o1 = o0 + 16;
    float v0 = alpha * acc0[r];
    float v1 = alpha * acc1[r];
    C[o0] = accumulate ? (C[o0] + v0) : v0;
    C[o1] = accumulate ? (C[o1] + v1) : v1;
  }
}

/* ------------------- pack x (B,T,C) -> sig f16 (N, T) --------------------- */
__global__ void k_pack_sig(const float* __restrict__ x, _Float16* __restrict__ sigH) {
  size_t i = (size_t)blockIdx.x * blockDim.x + threadIdx.x;
  if (i >= (size_t)WG_B * SEQ_T * CH_C) return;
  int c = (int)(i % CH_C);
  int t = (int)((i / CH_C) % SEQ_T);
  int b = (int)(i / ((size_t)CH_C * SEQ_T));
  sigH[((size_t)(b * CH_C + c)) * SEQ_T + t] = (_Float16)x[i];
}

/* ------- trig basis (f16): out[r][c] = cos/sin(2*pi*r*c/P), zero c>=Kv ---- */
__global__ void k_gen_dft(_Float16* __restrict__ cosO, _Float16* __restrict__ sinO,
                          int rows, int cols, int ld, int P, int Kv) {
  size_t i = (size_t)blockIdx.x * blockDim.x + threadIdx.x;
  if (i >= (size_t)rows * cols) return;
  int r = (int)(i / cols), c = (int)(i % cols);
  float sv = 0.f, cv = 0.f;
  if (c < Kv) {
    long m = ((long)r * (long)c) % P;
    float ang = 6.28318530717958647692f * (float)m / (float)P;
    __sincosf(ang, &sv, &cv);
  }
  cosO[(size_t)r * ld + c] = (_Float16)cv;
  sinO[(size_t)r * ld + c] = (_Float16)sv;
}

/* -------- FFT branch ------------------------------------------------------ */
__global__ void k_ampB(const float* __restrict__ Re, const float* __restrict__ Im,
                       float* __restrict__ ampB) {
  int i = blockIdx.x * blockDim.x + threadIdx.x;
  if (i >= WG_B * F_R) return;
  int b = i / F_R, f = i % F_R;
  float s = 0.f;
  for (int c = 0; c < CH_C; ++c) {
    size_t o = ((size_t)(b * CH_C + c)) * SEQ_T + f;
    float re = Re[o], im = Im[o];
    s += sqrtf(re * re + im * im);
  }
  ampB[i] = s * (1.f / CH_C);
}

__global__ void k_freqlist(const float* __restrict__ ampB, float* __restrict__ fl) {
  int f = blockIdx.x * blockDim.x + threadIdx.x;
  if (f >= F_R) return;
  if (f == 0) { fl[0] = 0.f; return; }
  float s = 0.f;
  for (int b = 0; b < WG_B; ++b) s += ampB[b * F_R + f];
  fl[f] = s * (1.f / WG_B);
}

__global__ __launch_bounds__(256)
void k_fft_top(const float* __restrict__ fl, const float* __restrict__ ampB,
               int* __restrict__ fftP, float* __restrict__ fftW) {
  __shared__ int sIdx[TOPK];
  if (threadIdx.x == 0) {
    int ch[TOPK];
    for (int p = 0; p < TOPK; ++p) {
      float bv = -3.4e38f; int bi = 0;
      for (int i = 0; i < F_R; ++i) {
        bool used = false;
        for (int q = 0; q < p; ++q) used |= (ch[q] == i);
        if (!used && fl[i] > bv) { bv = fl[i]; bi = i; }
      }
      ch[p] = bi; sIdx[p] = bi;
      fftP[p] = SEQ_T / (bi < 1 ? 1 : bi);
    }
  }
  __syncthreads();
  if (threadIdx.x < WG_B * TOPK) {
    int b = threadIdx.x / TOPK, i = threadIdx.x % TOPK;
    fftW[threadIdx.x] = ampB[b * F_R + sIdx[i]];
  }
}

/* -------- Wavelet branch -------------------------------------------------- */
__global__ __launch_bounds__(128)
void k_dwt(const _Float16* __restrict__ sigH, _Float16* __restrict__ caH,
           float* __restrict__ eSumW) {
  const int n = blockIdx.x, tid = threadIdx.x;
  __shared__ float red[128];
  float e = 0.f;
  for (int j = tid; j < TC_W; j += 128) {
    float acc = 0.f;
    #pragma unroll
    for (int i = 0; i < 8; ++i) {
      int q = 2 * j - 6 + i;
      if (q < 0) q = -q - 1;
      if (q >= SEQ_T) q = 2 * SEQ_T - 1 - q;
      acc += (float)sigH[(size_t)n * SEQ_T + q] * DB4[7 - i];
    }
    caH[(size_t)n * KW_P + j] = (_Float16)acc;
    e += acc * acc;
  }
  for (int j = TC_W + tid; j < KW_P; j += 128)   /* K-pad zeros */
    caH[(size_t)n * KW_P + j] = (_Float16)0.f;
  red[tid] = e; __syncthreads();
  for (int s = 64; s > 0; s >>= 1) {
    if (tid < s) red[tid] += red[tid + s];
    __syncthreads();
  }
  if (tid == 0) atomicAdd(eSumW, red[0]);
}

__global__ void k_wav_top(const float* __restrict__ ReC, const float* __restrict__ ImC,
                          int* __restrict__ cntW) {
  int n = blockIdx.x * blockDim.x + threadIdx.x;
  if (n >= NSIG) return;
  float bv[TOPK] = {-1.f, -1.f, -1.f, -1.f, -1.f};
  int   bi[TOPK] = {1, 1, 1, 1, 1};
  for (int f = 1; f < FC_W; ++f) {
    size_t o = (size_t)n * NW_P + f;
    float re = ReC[o], im = ImC[o];
    float a = re * re + im * im;           /* monotone w.r.t. amplitude */
    if (a > bv[TOPK - 1]) {
      int p = TOPK - 1;
      while (p > 0 && a > bv[p - 1]) { bv[p] = bv[p - 1]; bi[p] = bi[p - 1]; --p; }
      bv[p] = a; bi[p] = f;
    }
  }
  for (int i = 0; i < TOPK; ++i) {
    int idx = bi[i] < 1 ? 1 : bi[i];
    int per = (TC_W / idx) * 2;
    if (per >= 0 && per < 2 * TC_W + 2) atomicAdd(&cntW[per], 1);
  }
}

__global__ void k_top5_int(const int* __restrict__ cnt, int L, int* __restrict__ outP) {
  if (threadIdx.x != 0 || blockIdx.x != 0) return;
  int ch[TOPK];
  for (int p = 0; p < TOPK; ++p) {
    int bv = -2147483647; int bi = 0;
    for (int i = 0; i < L; ++i) {
      bool used = false;
      for (int q = 0; q < p; ++q) used |= (ch[q] == i);
      if (!used && cnt[i] > bv) { bv = cnt[i]; bi = i; }
    }
    ch[p] = bi; outP[p] = bi;
  }
}

/* -------- VMD: whole per-signal state lives in LDS (320KB/WGP) ------------ */
__global__ __launch_bounds__(256)
void k_vmd(const float* __restrict__ Re, const float* __restrict__ Im,
           _Float16* __restrict__ Ure, _Float16* __restrict__ Uim) {
  const int n = blockIdx.x, tid = threadIdx.x;
  __shared__ float fr[SEQ_T], fi[SEQ_T], lr[SEQ_T], li[SEQ_T];
  __shared__ float red[6 * 256];
  __shared__ float om[3];
  for (int t = tid; t < SEQ_T; t += 256) {
    int src = (t + SEQ_T / 2) & (SEQ_T - 1);        /* fftshift */
    fr[t] = Re[(size_t)n * SEQ_T + src];
    fi[t] = Im[(size_t)n * SEQ_T + src];
    lr[t] = 0.f; li[t] = 0.f;
  }
  if (tid < 3) om[tid] = 0.5f * hash01(n, tid);
  __syncthreads();
  const float TAU = 1e-7f, TAU2 = 1e-14f, ALPHA = 2000.f;
  for (int iter = 0; iter < 50; ++iter) {
    float o0 = om[0], o1 = om[1], o2 = om[2];
    float na0 = 0, na1 = 0, na2 = 0, da0 = 0, da1 = 0, da2 = 0;
    bool last = (iter == 49);
    for (int t = tid; t < SEQ_T; t += 256) {
      float f   = (float)t * (1.f / SEQ_T) - 0.5f;
      float fre = fr[t], fim = fi[t];
      float nr  = fre - 0.5f * lr[t], ni = fim - 0.5f * li[t];
      float e0 = (f - o0) * (f - o0);
      float e1 = (f - o1) * (f - o1);
      float e2 = (f - o2) * (f - o2);
      float i0 = 1.f / (1.f + ALPHA * (e0 + TAU2 + e1));
      float i1 = 1.f / (1.f + ALPHA * (e1 + TAU2 + e0 + e2));
      float i2 = 1.f / (1.f + ALPHA * (e2 + TAU2 + e1));
      float u0r = nr * i0, u0i = ni * i0;
      float u1r = nr * i1, u1i = ni * i1;
      float u2r = nr * i2, u2i = ni * i2;
      float p0 = u0r * u0r + u0i * u0i;
      float p1 = u1r * u1r + u1i * u1i;
      float p2 = u2r * u2r + u2i * u2i;
      na0 += f * p0; da0 += p0;
      na1 += f * p1; da1 += p1;
      na2 += f * p2; da2 += p2;
      lr[t] += TAU * ((u0r + u1r + u2r) - fre);
      li[t] += TAU * ((u0i + u1i + u2i) - fim);
      if (last) {
        size_t b0 = ((size_t)n * 3 + 0) * SEQ_T + t;
        size_t b1 = ((size_t)n * 3 + 1) * SEQ_T + t;
        size_t b2 = ((size_t)n * 3 + 2) * SEQ_T + t;
        Ure[b0] = (_Float16)u0r; Uim[b0] = (_Float16)u0i;
        Ure[b1] = (_Float16)u1r; Uim[b1] = (_Float16)u1i;
        Ure[b2] = (_Float16)u2r; Uim[b2] = (_Float16)u2i;
      }
    }
    red[tid] = na0; red[256 + tid] = na1; red[512 + tid] = na2;
    red[768 + tid] = da0; red[1024 + tid] = da1; red[1280 + tid] = da2;
    __syncthreads();
    for (int s = 128; s > 0; s >>= 1) {
      if (tid < s)
        for (int q = 0; q < 6; ++q) red[q * 256 + tid] += red[q * 256 + tid + s];
      __syncthreads();
    }
    if (tid < 3) om[tid] = red[tid * 256] / fmaxf(red[(3 + tid) * 256], 1e-30f);
    __syncthreads();
  }
}

/* imf post: energies + (-1)^t ifftshift sign into f16 for rfft GEMM */
__global__ __launch_bounds__(256)
void k_imf_post(const float* __restrict__ Imf, _Float16* __restrict__ imfH,
                float* __restrict__ energies) {
  const int row = blockIdx.x, tid = threadIdx.x;
  __shared__ float red[256];
  float e = 0.f;
  for (int t = tid; t < SEQ_T; t += 256) {
    float v = Imf[(size_t)row * SEQ_T + t];
    e += v * v;
    imfH[(size_t)row * SEQ_T + t] = (_Float16)((t & 1) ? -v : v);
  }
  red[tid] = e; __syncthreads();
  for (int s = 128; s > 0; s >>= 1) {
    if (tid < s) red[tid] += red[tid + s];
    __syncthreads();
  }
  if (tid == 0) energies[row] = red[0];
}

__global__ void k_top2(const float* __restrict__ energies, int* __restrict__ topi,
                       float* __restrict__ esum) {
  int n = blockIdx.x * blockDim.x + threadIdx.x;
  if (n >= NSIG) return;
  float e[3] = { energies[n * 3], energies[n * 3 + 1], energies[n * 3 + 2] };
  int i0 = 0;
  if (e[1] > e[i0]) i0 = 1;
  if (e[2] > e[i0]) i0 = 2;
  int i1 = -1; float b = -3.4e38f;
  for (int j = 0; j < 3; ++j)
    if (j != i0 && e[j] > b) { b = e[j]; i1 = j; }
  topi[2 * n] = i0; topi[2 * n + 1] = i1;
  atomicAdd(esum, e[i0] + e[i1]);
}

__global__ void k_gather_sel(const _Float16* __restrict__ imfH,
                             const int* __restrict__ topi,
                             _Float16* __restrict__ selH) {
  size_t i = (size_t)blockIdx.x * blockDim.x + threadIdx.x;
  if (i >= (size_t)2 * NSIG * SEQ_T) return;
  int t = (int)(i & (SEQ_T - 1));
  int row = (int)(i >> 11);
  int n = row >> 1, j = row & 1;
  int src = n * 3 + topi[2 * n + j];
  selH[i] = imfH[(size_t)src * SEQ_T + t];
}

__global__ void k_vmd_argmax(const float* __restrict__ SRe,
                             const float* __restrict__ SIm,
                             int* __restrict__ cntV) {
  int r = blockIdx.x * blockDim.x + threadIdx.x;
  if (r >= 2 * NSIG) return;
  int bi = 0; float bv = 0.f;
  for (int f = 1; f < F_R; ++f) {
    size_t o = (size_t)r * NS_P + f;
    float re = SRe[o], im = SIm[o];
    float a = re * re + im * im;
    if (a > bv) { bv = a; bi = f; }
  }
  int fidx = bi < 1 ? 1 : bi;
  int per = SEQ_T / fidx;
  if (per >= 0 && per <= SEQ_T) atomicAdd(&cntV[per], 1);
}

/* -------- fusion ---------------------------------------------------------- */
__global__ __launch_bounds__(256)
void k_fuse(const int* __restrict__ fftP, const float* __restrict__ fftW,
            const int* __restrict__ wavP, const float* __restrict__ eSumW,
            const int* __restrict__ vmdP, const float* __restrict__ vmdE,
            float* __restrict__ out) {
  __shared__ int cnt[4 * SEQ_T];
  __shared__ int fp[TOPK];
  __shared__ float wS, vS;
  int tid = threadIdx.x;
  for (int i = tid; i < 4 * SEQ_T; i += 256) cnt[i] = 0;
  __syncthreads();
  if (tid == 0) {
    for (int i = 0; i < TOPK; ++i) {
      int p;
      p = fftP[i]; p = p < 0 ? 0 : (p > 4 * SEQ_T - 1 ? 4 * SEQ_T - 1 : p); cnt[p]++;
      p = wavP[i]; p = p < 0 ? 0 : (p > 4 * SEQ_T - 1 ? 4 * SEQ_T - 1 : p); cnt[p]++;
      p = vmdP[i]; p = p < 0 ? 0 : (p > 4 * SEQ_T - 1 ? 4 * SEQ_T - 1 : p); cnt[p]++;
    }
    int ch[TOPK];
    for (int pk = 0; pk < TOPK; ++pk) {
      int bv = -2147483647; int bi = 0;
      for (int i = 0; i < 4 * SEQ_T; ++i) {
        int v = (i > 1 && i < SEQ_T / 2) ? cnt[i] : -1;
        bool used = false;
        for (int q = 0; q < pk; ++q) used |= (ch[q] == i);
        if (!used && v > bv) { bv = v; bi = i; }
      }
      ch[pk] = bi; fp[pk] = bi;
    }
    wS = (eSumW[0] / (float)NSIG) * 0.2f;
    vS = (vmdE[0] / (float)(2 * NSIG)) * 0.2f;
  }
  __syncthreads();
  if (tid < TOPK) out[tid] = (float)fp[tid];
  if (tid < WG_B) {
    float fb = 5.f * wS + 5.f * vS;
    for (int i = 0; i < TOPK; ++i) fb += fftW[tid * TOPK + i];
    fb *= (1.f / 15.f);
    for (int i = 0; i < TOPK; ++i) {
      int p = fp[i];
      float acc = 0.f, np = 0.f;
      for (int j = 0; j < TOPK; ++j)
        if (fftP[j] == p) { acc += fftW[tid * TOPK + j]; np += 1.f; break; }
      for (int j = 0; j < TOPK; ++j)
        if (wavP[j] == p) { acc += wS; np += 1.f; break; }
      for (int j = 0; j < TOPK; ++j)
        if (vmdP[j] == p) { acc += vS; np += 1.f; break; }
      out[TOPK + tid * TOPK + i] = (np > 0.f) ? acc / np : fb;
    }
  }
}

/* ========================= host orchestration ============================ */
extern "C" void kernel_launch(void* const* d_in, const int* in_sizes, int n_in,
                              void* d_out, int out_size, void* d_ws, size_t ws_size,
                              hipStream_t stream)
{
  (void)in_sizes; (void)n_in; (void)out_size; (void)ws_size;
  const float* x = (const float*)d_in[0];
  float* out = (float*)d_out;

  char* base = (char*)d_ws;
  size_t off = 0;
  auto alloc = [&](size_t bytes) -> void* {
    void* p = base + off;
    off = (off + bytes + 255) & ~(size_t)255;
    return p;
  };

  const size_t NT = (size_t)NSIG * SEQ_T;
  _Float16* sigH = (_Float16*)alloc(NT * 2);
  _Float16* cosM = (_Float16*)alloc(NT * 2);   /* symmetric: also its own ^T */
  _Float16* sinM = (_Float16*)alloc(NT * 2);   /* symmetric: also its own ^T */
  float*    Re   = (float*)alloc(NT * 4);
  float*    Im   = (float*)alloc(NT * 4);
  float*    ampB = (float*)alloc((size_t)WG_B * F_R * 4);
  float*    frqL = (float*)alloc((size_t)F_R * 4);
  int*      fftP = (int*)alloc(TOPK * 4);
  float*    fftW = (float*)alloc((size_t)WG_B * TOPK * 4);
  _Float16* caH  = (_Float16*)alloc((size_t)NSIG * KW_P * 2);
  float*    eSumW= (float*)alloc(4);
  _Float16* cos2T= (_Float16*)alloc((size_t)NW_P * KW_P * 2);  /* [f][t], K-pad 0 */
  _Float16* sin2T= (_Float16*)alloc((size_t)NW_P * KW_P * 2);
  float*    ReC  = (float*)alloc((size_t)NSIG * NW_P * 4);
  float*    ImC  = (float*)alloc((size_t)NSIG * NW_P * 4);
  int*      cntW = (int*)alloc(2560 * 4);
  int*      wavP = (int*)alloc(TOPK * 4);
  _Float16* Ure  = (_Float16*)alloc(NT * 3 * 2);
  _Float16* Uim  = (_Float16*)alloc(NT * 3 * 2);
  float*    Imf  = (float*)alloc(NT * 3 * 4);
  _Float16* imfH = (_Float16*)alloc(NT * 3 * 2);
  float*    ener = (float*)alloc((size_t)NSIG * 3 * 4);
  int*      topi = (int*)alloc((size_t)NSIG * 2 * 4);
  float*    vmdE = (float*)alloc(4);
  int*      cntV = (int*)alloc(2560 * 4);
  int*      vmdP = (int*)alloc(TOPK * 4);
  /* aliases on dead buffers (strictly stream-ordered):                     */
  _Float16* selH = (_Float16*)Re;               /* 16 MiB in 16 MiB          */
  float*    SRe  = Imf;                         /* 17.8 MiB                  */
  float*    SIm  = (float*)((char*)Imf +
                   (((size_t)2 * NSIG * NS_P * 4 + 255) & ~(size_t)255));

  (void)hipMemsetAsync(cntW, 0, 2560 * 4, stream);
  (void)hipMemsetAsync(cntV, 0, 2560 * 4, stream);
  (void)hipMemsetAsync(eSumW, 0, 4, stream);
  (void)hipMemsetAsync(vmdE, 0, 4, stream);

  dim3 b256(256);
  auto gemm = [&](const _Float16* A, int lda, const _Float16* BT, int ldbt,
                  float* C, int ldc, int M, int N, int K, float alpha, int acc) {
    dim3 g((unsigned)(N / 64), (unsigned)(M / 64));
    k_wmma_gemm<<<g, b256, 0, stream>>>(A, lda, BT, ldbt, C, ldc, M, N, K, alpha, acc);
  };

  /* stage 0: pack + trig bases */
  {
    size_t tot = (size_t)WG_B * SEQ_T * CH_C;
    k_pack_sig<<<dim3((unsigned)((tot + 255) / 256)), b256, 0, stream>>>(x, sigH);
    size_t tg = (size_t)SEQ_T * SEQ_T;
    k_gen_dft<<<dim3((unsigned)((tg + 255) / 256)), b256, 0, stream>>>(
        cosM, sinM, SEQ_T, SEQ_T, SEQ_T, SEQ_T, SEQ_T);
    size_t t2 = (size_t)NW_P * KW_P;   /* transposed wavelet basis [f][t] */
    k_gen_dft<<<dim3((unsigned)((t2 + 255) / 256)), b256, 0, stream>>>(
        cos2T, sin2T, NW_P, KW_P, KW_P, TC_W, TC_W);
  }

  /* stage 1: full complex DFT of all signals (WMMA) */
  gemm(sigH, SEQ_T, cosM, SEQ_T, Re, SEQ_T, NSIG, SEQ_T, SEQ_T, 1.f, 0);
  gemm(sigH, SEQ_T, sinM, SEQ_T, Im, SEQ_T, NSIG, SEQ_T, SEQ_T, -1.f, 0);

  /* stage 2: FFT branch */
  k_ampB<<<dim3((WG_B * F_R + 255) / 256), b256, 0, stream>>>(Re, Im, ampB);
  k_freqlist<<<dim3((F_R + 255) / 256), b256, 0, stream>>>(ampB, frqL);
  k_fft_top<<<dim3(1), b256, 0, stream>>>(frqL, ampB, fftP, fftW);

  /* stage 3: wavelet branch (K padded to 1056, N padded to 576) */
  k_dwt<<<dim3(NSIG), dim3(128), 0, stream>>>(sigH, caH, eSumW);
  gemm(caH, KW_P, cos2T, KW_P, ReC, NW_P, NSIG, NW_P, KW_P, 1.f, 0);
  gemm(caH, KW_P, sin2T, KW_P, ImC, NW_P, NSIG, NW_P, KW_P, -1.f, 0);
  k_wav_top<<<dim3((NSIG + 255) / 256), b256, 0, stream>>>(ReC, ImC, cntW);
  k_top5_int<<<dim3(1), dim3(1), 0, stream>>>(cntW, 2 * TC_W + 2, wavP);

  /* stage 4: VMD (LDS-resident) + ifft via WMMA */
  k_vmd<<<dim3(NSIG), b256, 0, stream>>>(Re, Im, Ure, Uim);
  const float invT = 1.f / (float)SEQ_T;
  gemm(Ure, SEQ_T, cosM, SEQ_T, Imf, SEQ_T, 3 * NSIG, SEQ_T, SEQ_T, invT, 0);
  gemm(Uim, SEQ_T, sinM, SEQ_T, Imf, SEQ_T, 3 * NSIG, SEQ_T, SEQ_T, -invT, 1);
  k_imf_post<<<dim3(3 * NSIG), b256, 0, stream>>>(Imf, imfH, ener);
  k_top2<<<dim3((NSIG + 255) / 256), b256, 0, stream>>>(ener, topi, vmdE);
  {
    size_t tot = (size_t)2 * NSIG * SEQ_T;
    k_gather_sel<<<dim3((unsigned)((tot + 255) / 256)), b256, 0, stream>>>(imfH, topi, selH);
  }
  gemm(selH, SEQ_T, cosM, SEQ_T, SRe, NS_P, 2 * NSIG, NS_P, SEQ_T, 1.f, 0);
  gemm(selH, SEQ_T, sinM, SEQ_T, SIm, NS_P, 2 * NSIG, NS_P, SEQ_T, -1.f, 0);
  k_vmd_argmax<<<dim3((2 * NSIG + 255) / 256), b256, 0, stream>>>(SRe, SIm, cntV);
  k_top5_int<<<dim3(1), dim3(1), 0, stream>>>(cntV, SEQ_T + 1, vmdP);

  /* stage 5: fusion */
  k_fuse<<<dim3(1), b256, 0, stream>>>(fftP, fftW, wavP, eSumW, vmdP, vmdE, out);
}
